// Model_17428977287605
// MI455X (gfx1250) — 8-bit matrix operand order repaired and checked by reading, compiled, not run on hardware
//
#include <hip/hip_runtime.h>
#include <stdint.h>

#define NI 32
#define CI 256
#define HI 56
#define WI 56
#define HP 58
#define WP 58
#define SPI (HI*WI)
#define SPTOT (NI*SPI)
#define KRED (CI*9)
#define KSTEPS 36
#define NCOLS 32
#define LDS_COL_STRIDE 80
#define LDS_STEP (NCOLS*LDS_COL_STRIDE)

typedef __attribute__((ext_vector_type(8))) int v8i;

union V8 {
  v8i   v;
  int   i[8];
  uint2 u2[4];
  uint4 u4[2];
};

__global__ void quant_input_pad(const float* __restrict__ x, uint8_t* __restrict__ xp) {
  int idx = blockIdx.x * blockDim.x + threadIdx.x;
  const int total = NI * HP * WP * (CI / 4);
  if (idx >= total) return;
  int c4 = idx & 63;
  int t  = idx >> 6;
  int wp = t % WP; t /= WP;
  int hp = t % HP;
  int n  = t / HP;
  uint32_t pack = 0u;
  if (hp >= 1 && hp <= HI && wp >= 1 && wp <= WI) {
    int h = hp - 1, w = wp - 1;
    const float* px = x + (((size_t)n * CI + c4 * 4) * HI + h) * WI + w;
#pragma unroll
    for (int j = 0; j < 4; ++j) {
      float f = px[(size_t)j * SPI];
      int q = (int)rintf(f * 128.0f);
      q = q < 0 ? 0 : (q > 255 ? 255 : q);
      pack |= ((uint32_t)q) << (8 * j);
    }
  }
  *(uint32_t*)(xp + (size_t)idx * 4) = pack;
}

__global__ void quant_weights(const float* __restrict__ w, int8_t* __restrict__ wq) {
  int idx = blockIdx.x * blockDim.x + threadIdx.x;
  const int total = CI * 9 * (CI / 4);
  if (idx >= total) return;
  int c4 = idx & 63;
  int rs = (idx >> 6) % 9;
  int o  = idx / (64 * 9);
  int r = rs / 3, s = rs - r * 3;
  uint32_t pack = 0u;
#pragma unroll
  for (int j = 0; j < 4; ++j) {
    int c = c4 * 4 + j;
    float f = w[(((size_t)o * CI + c) * 3 + r) * 3 + s];
    int q = (int)rintf(f);
    q = q < -127 ? -127 : (q > 127 ? 127 : q);
    pack |= ((uint32_t)(uint8_t)(int8_t)q) << (8 * j);
  }
  *(uint32_t*)((uint8_t*)wq + (size_t)idx * 4) = pack;
}

template<bool FUSE>
__global__ __launch_bounds__(128)
void conv3x3_iu8_wmma(const uint8_t* __restrict__ act,
                      const int8_t*  __restrict__ wq,
                      uint8_t*       __restrict__ out_q,
                      const float*   __restrict__ xres,
                      float*         __restrict__ out_f)
{
  __shared__ uint8_t smem[KSTEPS * LDS_STEP];
  const int tid  = threadIdx.x;
  const int lane = tid & 31;
  const int wave = tid >> 5;
  const int sp_base = blockIdx.x * NCOLS;

  {
    const int col  = tid >> 2;
    const int part = tid & 3;
    int sp  = sp_base + col;
    int n   = sp / SPI;
    int rem = sp - n * SPI;
    int h   = rem / WI;
    int w   = rem - h * WI;
    const uint8_t* gp0 =
        act + (((size_t)(n * HP + h)) * WP + w) * CI + (size_t)part * 16;
    uint32_t l0 = (uint32_t)(uintptr_t)(void*)smem
                + (uint32_t)(col * LDS_COL_STRIDE + part * 16);
#pragma unroll 1
    for (int step = 0; step < KSTEPS; ++step) {
      int rs = step >> 2, cblk = step & 3;
      int r = rs / 3, s = rs - r * 3;
      uint64_t ga = (uint64_t)(uintptr_t)(gp0 + ((r * WP + s) * CI + cblk * 64));
      uint32_t la = l0 + (uint32_t)(step * LDS_STEP);
      asm volatile("global_load_async_to_lds_b128 %0, %1, off"
                   :: "v"(la), "v"(ga) : "memory");
    }
  }
  asm volatile("s_wait_asynccnt 0" ::: "memory");
  __syncthreads();

  V8 acc[4][2];
#pragma unroll
  for (int t = 0; t < 4; ++t)
#pragma unroll
    for (int u = 0; u < 2; ++u)
#pragma unroll
      for (int j = 0; j < 8; ++j) acc[t][u].i[j] = 0;

  const int sub8 = (lane >> 4) * 8;
  const uint8_t* arow =
      (const uint8_t*)wq + (size_t)(wave * 64 + (lane & 15)) * KRED + sub8;
  const int boff = (lane & 15) * LDS_COL_STRIDE + (lane >> 4) * 16;

#pragma unroll 1
  for (int step = 0; step < KSTEPS; ++step) {
    V8 b[2];
#pragma unroll
    for (int u = 0; u < 2; ++u) {
      const uint8_t* bp = smem + step * LDS_STEP + u * (16 * LDS_COL_STRIDE) + boff;
      b[u].u4[0] = *(const uint4*)(bp);
      b[u].u4[1] = *(const uint4*)(bp + 32);
    }
#pragma unroll
    for (int t = 0; t < 4; ++t) {
      V8 a;
      const uint8_t* ap = arow + (size_t)t * 16 * KRED + (size_t)step * 64;
#pragma unroll
      for (int j = 0; j < 4; ++j)
        a.u2[j] = *(const uint2*)(ap + sub8 + (j >> 1) * 32 + (j & 1) * 8);
#pragma unroll
      for (int u = 0; u < 2; ++u)
        acc[t][u].v = __builtin_amdgcn_wmma_i32_16x16x64_iu8(
            true, a.v, false, b[u].v, acc[t][u].v, false, false);
    }
  }

#pragma unroll
  for (int u = 0; u < 2; ++u) {
    int sp  = sp_base + u * 16 + (lane & 15);
    int n   = sp / SPI;
    int rem = sp - n * SPI;
    int h   = rem / WI;
    int w   = rem - h * WI;
#pragma unroll
    for (int t = 0; t < 4; ++t) {
      int k0 = wave * 64 + t * 16 + sub8;
      if (!FUSE) {
        uint32_t lo = 0, hi = 0;
#pragma unroll
        for (int j = 0; j < 8; ++j) {
          int q = (int)rintf((float)acc[t][u].i[j] * (1.0f / 128.0f));
          q = q < 0 ? 0 : (q > 255 ? 255 : q);
          if (j < 4) lo |= (uint32_t)q << (8 * j);
          else       hi |= (uint32_t)q << (8 * (j - 4));
        }
        size_t off = (((size_t)(n * HP + h + 1)) * WP + (w + 1)) * CI + k0;
        *(uint2*)(out_q + off) = make_uint2(lo, hi);
      } else {
#pragma unroll
        for (int j = 0; j < 8; ++j) {
          int k = k0 + j;
          size_t idx = (((size_t)n * CI + k) * HI + h) * WI + w;
          float rr  = (float)acc[t][u].i[j] * (1.0f / 16384.0f);
          float sum = (rr + xres[idx]) * 16384.0f;
          sum = fminf(sum,  2147483647.0f);
          sum = fmaxf(sum, -2147483647.0f);
          float o = sum * (1.0f / 16384.0f);
          out_f[idx] = fmaxf(o, 0.0f);
        }
      }
    }
  }
}

extern "C" void kernel_launch(void* const* d_in, const int* in_sizes, int n_in,
                              void* d_out, int out_size, void* d_ws, size_t ws_size,
                              hipStream_t stream) {
  const float* x  = (const float*)d_in[0];
  const float* w1 = (const float*)d_in[1];
  const float* w2 = (const float*)d_in[2];

  uint8_t* ws = (uint8_t*)d_ws;
  const size_t XP_BYTES = (size_t)NI * HP * WP * CI;
  const size_t WQ_BYTES = (size_t)CI * 9 * CI;
  uint8_t* xp  = ws;
  uint8_t* r1p = ws + XP_BYTES;
  int8_t*  w1q = (int8_t*)(ws + 2 * XP_BYTES);
  int8_t*  w2q = (int8_t*)(ws + 2 * XP_BYTES + WQ_BYTES);

  hipMemsetAsync(r1p, 0, XP_BYTES, stream);

  {
    int total = NI * HP * WP * (CI / 4);
    quant_input_pad<<<(total + 255) / 256, 256, 0, stream>>>(x, xp);
  }
  {
    int total = CI * 9 * (CI / 4);
    quant_weights<<<(total + 255) / 256, 256, 0, stream>>>(w1, w1q);
    quant_weights<<<(total + 255) / 256, 256, 0, stream>>>(w2, w2q);
  }

  conv3x3_iu8_wmma<false><<<SPTOT / NCOLS, 128, 0, stream>>>(
      xp, w1q, r1p, nullptr, nullptr);
  conv3x3_iu8_wmma<true><<<SPTOT / NCOLS, 128, 0, stream>>>(
      r1p, w2q, nullptr, x, (float*)d_out);
}
